// TimeStyleSpaceMapper_84456236909290
// MI455X (gfx1250) — compile-verified, matrix operand order checked by reading, not run on hardware
//
#include <hip/hip_runtime.h>
#include <math.h>

typedef __attribute__((ext_vector_type(16))) _Float16 v16h;
typedef __attribute__((ext_vector_type(8)))  float    v8f;

#define NLAYERS 26
#define B_ROWS  8192
#define XCOLS   9088
#define CTXK    544      // 515 padded to multiple of 32
#define CTXREAL 515
#define BM      128
#define BN      32
#define BK      32
#define LDA     40       // LDS row stride in halves (80B, 16B-aligned chunks)

union V16 { v16h h; int4 i[2]; };

typedef __attribute__((address_space(3))) _Float16 lds_h;

__device__ __forceinline__ unsigned lds_off(_Float16* p) {
    return (unsigned)(__SIZE_TYPE__)(lds_h*)p;
}

// ---- CDNA5 async memory->LDS staging (ASYNCcnt-tracked), GVS addressing ----
__device__ __forceinline__ void async_b64(unsigned lds, const void* base, int byteOff) {
    asm volatile("global_load_async_to_lds_b64 %0, %1, %2"
                 :: "v"(lds), "v"(byteOff), "s"(base) : "memory");
}
__device__ __forceinline__ void async_b128(unsigned lds, const void* base, int byteOff) {
    asm volatile("global_load_async_to_lds_b128 %0, %1, %2"
                 :: "v"(lds), "v"(byteOff), "s"(base) : "memory");
}
#define ASYNC_WAIT0() asm volatile("s_wait_asynccnt 0x0" ::: "memory")

#define WMMA_F16(A_, B_, C_) \
    __builtin_amdgcn_wmma_f32_16x16x32_f16(false, (A_).h, false, (B_).h, (short)0, (C_), false, false)

__device__ __forceinline__ float flrelu(float v) {
    return (v > 0.f ? v : 0.2f * v) * 1.41421356237309515f;
}
__device__ __forceinline__ float fsigmoid(float v) {
    return 1.f / (1.f + __expf(-v));
}

// ---------------- ctx_emb = [beta, sin(beta), cos(beta), context] -> f16, padded to 544
__global__ void build_ctx_kernel(const float* __restrict__ beta,
                                 const float* __restrict__ context,
                                 _Float16* __restrict__ out) {
    int idx = blockIdx.x * 256 + threadIdx.x;
    if (idx >= B_ROWS * CTXK) return;
    int r = idx / CTXK, k = idx % CTXK;
    float v;
    if (k == 0)            v = beta[r];
    else if (k == 1)       v = sinf(beta[r]);
    else if (k == 2)       v = cosf(beta[r]);
    else if (k < CTXREAL)  v = context[r * 512 + (k - 3)];
    else                   v = 0.f;
    out[idx] = (_Float16)v;
}

// ---------------- weight conversions (prescaled by EqualLinear scale) ----------------
__global__ void conv_scale_kernel(const float* __restrict__ src, _Float16* __restrict__ dst,
                                  int n, float s) {
    int i = blockIdx.x * 256 + threadIdx.x;
    if (i < n) dst[i] = (_Float16)(src[i] * s);
}

__global__ void conv_pad_kernel(const float* __restrict__ src, _Float16* __restrict__ dst,
                                int rows, float s) {
    int i = blockIdx.x * 256 + threadIdx.x;
    if (i >= rows * CTXK) return;
    int r = i / CTXK, k = i % CTXK;
    dst[i] = (k < CTXREAL) ? (_Float16)(src[r * CTXREAL + k] * s) : (_Float16)0.f;
}

// ---------------- fused per-layer kernel ----------------
// out = lrelu(x Wl^T) * sigmoid(lrelu(ctx Wg^T)) + lrelu(ctx Wb^T)
__global__ __launch_bounds__(256) void fused_layer_kernel(
    const float*    __restrict__ x,     int xColOff,
    const _Float16* __restrict__ ctx16,
    const _Float16* __restrict__ Wl,
    const _Float16* __restrict__ Wg,
    const _Float16* __restrict__ Wb,
    const float*    __restrict__ pbl,
    const float*    __restrict__ pbg,
    const float*    __restrict__ pbb,
    float*          __restrict__ out,   int c)
{
    __shared__ __align__(16) _Float16 As [2][BM * LDA];
    __shared__ __align__(16) _Float16 Bs0[2][BN * LDA];
    __shared__ __align__(16) _Float16 Bs1[2][BN * LDA];

    const int tid   = threadIdx.x;
    const int lane  = tid & 31;
    const int wave  = tid >> 5;
    const int waveM = wave & 3;   // 0..3 -> 32-row stripes
    const int waveN = wave >> 2;  // 0..1 -> 16-col stripes
    const int mBlock = blockIdx.x * BM;
    const int nBlock = blockIdx.y * BN;

    const int lhalf = lane & 15;
    const int khA   = (lane < 16) ? 0 : 8;   // CDNA5 16-bit operand K-chunk base

    // staging thread mapping
    const int xr = tid >> 1, xk = (tid & 1) * 16;   // A tile: 128x32, 16 elems/thread
    const int br = tid >> 3, bk = (tid & 7) * 4;    // B tile: 32x32,   4 halves/thread

    const float* xsrc = x + (size_t)(mBlock + xr) * XCOLS + xColOff + xk;

    // per-thread LDS byte addresses for async destinations
    const unsigned b0Dst[2] = { lds_off(&Bs0[0][br * LDA + bk]), lds_off(&Bs0[1][br * LDA + bk]) };
    const unsigned b1Dst[2] = { lds_off(&Bs1[0][br * LDA + bk]), lds_off(&Bs1[1][br * LDA + bk]) };
    const unsigned aDst[2][2] = {
        { lds_off(&As[0][xr * LDA + xk]), lds_off(&As[0][xr * LDA + xk + 8]) },
        { lds_off(&As[1][xr * LDA + xk]), lds_off(&As[1][xr * LDA + xk + 8]) }
    };

    // per-wave fragment element offsets (constant-buffer indexing keeps LDS addrspace)
    const int aOff = (waveM * 32 + lhalf) * LDA;
    const int bOff = (waveN * 16 + lhalf) * LDA;
    const int sOff = xr * LDA + xk;

    v8f accL0 = {}, accL1 = {};
    v8f accG0 = {}, accG1 = {};
    v8f accB0 = {}, accB1 = {};

    // ================= pass 1: layer GEMM over K = c =================
    const int wlByte0 = ((nBlock + br) * c + bk) * 2;
    const int nIter1 = c / BK;

    async_b64(b0Dst[0], Wl, wlByte0);                       // B tile k0=0
    float4 fA0, fA1, fA2, fA3, fB0, fB1, fB2, fB3;
    fA0 = ((const float4*)xsrc)[0];
    fA1 = ((const float4*)xsrc)[1];
    fA2 = ((const float4*)xsrc)[2];
    fA3 = ((const float4*)xsrc)[3];

// consume register set C*, preload register set N* (no rotation moves)
#define P1_STEP(I_, BUF_, C0,C1,C2,C3, N0,N1,N2,N3)                              \
    {                                                                            \
        const bool nxt = ((I_) + 1) < nIter1;                                    \
        if (nxt) {                                                               \
            const float* s_ = xsrc + ((I_) + 1) * BK;                            \
            __builtin_prefetch(s_ + BK, 0, 1);                                   \
            N0 = ((const float4*)s_)[0]; N1 = ((const float4*)s_)[1];            \
            N2 = ((const float4*)s_)[2]; N3 = ((const float4*)s_)[3];            \
        }                                                                        \
        V16 v;                                                                   \
        v.h[0]=(_Float16)C0.x;  v.h[1]=(_Float16)C0.y;                           \
        v.h[2]=(_Float16)C0.z;  v.h[3]=(_Float16)C0.w;                           \
        v.h[4]=(_Float16)C1.x;  v.h[5]=(_Float16)C1.y;                           \
        v.h[6]=(_Float16)C1.z;  v.h[7]=(_Float16)C1.w;                           \
        v.h[8]=(_Float16)C2.x;  v.h[9]=(_Float16)C2.y;                           \
        v.h[10]=(_Float16)C2.z; v.h[11]=(_Float16)C2.w;                          \
        v.h[12]=(_Float16)C3.x; v.h[13]=(_Float16)C3.y;                          \
        v.h[14]=(_Float16)C3.z; v.h[15]=(_Float16)C3.w;                          \
        ASYNC_WAIT0();                                                           \
        int4* d_ = (int4*)(&As[BUF_][sOff]);                                     \
        d_[0] = v.i[0]; d_[1] = v.i[1];                                          \
        __syncthreads();                                                         \
        if (nxt) async_b64(b0Dst[1 - (BUF_)], Wl, wlByte0 + ((I_) + 1) * BK * 2);\
        V16 bf_, a0_, a1_;                                                       \
        { const _Float16* p_ = &Bs0[BUF_][bOff];                                 \
          bf_.i[0] = *(const int4*)(p_ + khA);                                   \
          bf_.i[1] = *(const int4*)(p_ + khA + 16); }                            \
        { const _Float16* p_ = &As[BUF_][aOff];                                  \
          a0_.i[0] = *(const int4*)(p_ + khA);                                   \
          a0_.i[1] = *(const int4*)(p_ + khA + 16);                              \
          a1_.i[0] = *(const int4*)(p_ + 16 * LDA + khA);                        \
          a1_.i[1] = *(const int4*)(p_ + 16 * LDA + khA + 16); }                 \
        accL0 = WMMA_F16(a0_, bf_, accL0);                                       \
        accL1 = WMMA_F16(a1_, bf_, accL1);                                       \
    }

    for (int i = 0; i < nIter1; i += 2) {
        P1_STEP(i, 0, fA0, fA1, fA2, fA3, fB0, fB1, fB2, fB3);
        if (i + 1 < nIter1) P1_STEP(i + 1, 1, fB0, fB1, fB2, fB3, fA0, fA1, fA2, fA3);
    }
#undef P1_STEP

    __syncthreads();   // all pass-1 reads done before pass-2 async writes

    // ================= pass 2: gate + bias GEMMs over K = 544 =================
    const int ctxByte0 = (int)(((size_t)(mBlock + xr) * CTXK + xk) * 2);
    const int wgByte0  = ((nBlock + br) * CTXK + bk) * 2;
    const int nIter2   = CTXK / BK;   // 17

    // prologue: tile 0 into buffer 0 (fully async: ctx A tile + both B tiles)
    async_b128(aDst[0][0], ctx16, ctxByte0);
    async_b128(aDst[0][1], ctx16, ctxByte0 + 16);
    async_b64 (b0Dst[0],  Wg,    wgByte0);
    async_b64 (b1Dst[0],  Wb,    wgByte0);

#define P2_STEP(I_, BUF_)                                                        \
    {                                                                            \
        ASYNC_WAIT0();                                                           \
        __syncthreads();                                                         \
        if ((I_) + 1 < nIter2) {                                                 \
            const int kb_ = ((I_) + 1) * BK * 2;                                 \
            async_b128(aDst[1 - (BUF_)][0], ctx16, ctxByte0 + kb_);              \
            async_b128(aDst[1 - (BUF_)][1], ctx16, ctxByte0 + kb_ + 16);         \
            async_b64 (b0Dst[1 - (BUF_)],  Wg,    wgByte0 + kb_);                \
            async_b64 (b1Dst[1 - (BUF_)],  Wb,    wgByte0 + kb_);                \
        }                                                                        \
        V16 bg_, bb_, a0_, a1_;                                                  \
        { const _Float16* p_ = &Bs0[BUF_][bOff];                                 \
          bg_.i[0] = *(const int4*)(p_ + khA);                                   \
          bg_.i[1] = *(const int4*)(p_ + khA + 16); }                            \
        { const _Float16* p_ = &Bs1[BUF_][bOff];                                 \
          bb_.i[0] = *(const int4*)(p_ + khA);                                   \
          bb_.i[1] = *(const int4*)(p_ + khA + 16); }                            \
        { const _Float16* p_ = &As[BUF_][aOff];                                  \
          a0_.i[0] = *(const int4*)(p_ + khA);                                   \
          a0_.i[1] = *(const int4*)(p_ + khA + 16);                              \
          a1_.i[0] = *(const int4*)(p_ + 16 * LDA + khA);                        \
          a1_.i[1] = *(const int4*)(p_ + 16 * LDA + khA + 16); }                 \
        accG0 = WMMA_F16(a0_, bg_, accG0);                                       \
        accG1 = WMMA_F16(a1_, bg_, accG1);                                       \
        accB0 = WMMA_F16(a0_, bb_, accB0);                                       \
        accB1 = WMMA_F16(a1_, bb_, accB1);                                       \
    }

    for (int i = 0; i < nIter2; i += 2) {
        P2_STEP(i, 0);
        if (i + 1 < nIter2) P2_STEP(i + 1, 1);
    }
#undef P2_STEP

    // ================= epilogue =================
    const int nloc = nBlock + waveN * 16 + lhalf;       // column within layer
    const float blv = pbl[nloc] * 0.01f;
    const float bgv = pbg[nloc] * 0.01f;
    const float bbv = pbb[nloc] * 0.01f;
    const int mBase = mBlock + waveM * 32 + ((lane < 16) ? 0 : 8);

#pragma unroll
    for (int j = 0; j < 8; ++j) {
        {
            float al = flrelu(accL0[j] + blv);
            float g  = fsigmoid(flrelu(accG0[j] + bgv));
            float bi = flrelu(accB0[j] + bbv);
            out[(size_t)(mBase + j) * XCOLS + xColOff + nloc] = al * g + bi;
        }
        {
            float al = flrelu(accL1[j] + blv);
            float g  = fsigmoid(flrelu(accG1[j] + bgv));
            float bi = flrelu(accB1[j] + bbv);
            out[(size_t)(mBase + 16 + j) * XCOLS + xColOff + nloc] = al * g + bi;
        }
    }
}

// ---------------- host launcher ----------------
static const int SS_DIMS[NLAYERS] = {
    512,512,512,512,512,512,512,512,512,512,512,512,512,512,512,
    256,256,256, 128,128,128, 64,64,64, 32,32
};

extern "C" void kernel_launch(void* const* d_in, const int* in_sizes, int n_in,
                              void* d_out, int out_size, void* d_ws, size_t ws_size,
                              hipStream_t stream) {
    (void)in_sizes; (void)n_in; (void)out_size; (void)ws_size;

    const float* x       = (const float*)d_in[0];
    const float* beta    = (const float*)d_in[1];
    const float* context = (const float*)d_in[2];
    float* out           = (float*)d_out;
    _Float16* ws         = (_Float16*)d_ws;

    size_t off = 0;
    _Float16* ctx16 = ws + off;
    off += (size_t)B_ROWS * CTXK;

    {
        int total = B_ROWS * CTXK;
        build_ctx_kernel<<<(total + 255) / 256, 256, 0, stream>>>(beta, context, ctx16);
    }

    _Float16* wlp[NLAYERS];
    _Float16* wgp[NLAYERS];
    _Float16* wbp[NLAYERS];
    const float sg = 0.01f / sqrtf((float)CTXREAL);

    for (int i = 0; i < NLAYERS; ++i) {
        int c = SS_DIMS[i];
        const float* Wl = (const float*)d_in[3 + 6 * i + 0];
        const float* Wg = (const float*)d_in[3 + 6 * i + 2];
        const float* Wb = (const float*)d_in[3 + 6 * i + 4];
        wlp[i] = ws + off; off += (size_t)c * c;
        wgp[i] = ws + off; off += (size_t)c * CTXK;
        wbp[i] = ws + off; off += (size_t)c * CTXK;
        float sl = 0.01f / sqrtf((float)c);
        conv_scale_kernel<<<(c * c + 255) / 256, 256, 0, stream>>>(Wl, wlp[i], c * c, sl);
        conv_pad_kernel<<<(c * CTXK + 255) / 256, 256, 0, stream>>>(Wg, wgp[i], c, sg);
        conv_pad_kernel<<<(c * CTXK + 255) / 256, 256, 0, stream>>>(Wb, wbp[i], c, sg);
    }

    int colOff = 0;
    for (int i = 0; i < NLAYERS; ++i) {
        int c = SS_DIMS[i];
        const float* pbl = (const float*)d_in[3 + 6 * i + 1];
        const float* pbg = (const float*)d_in[3 + 6 * i + 3];
        const float* pbb = (const float*)d_in[3 + 6 * i + 5];
        dim3 grid(B_ROWS / BM, c / BN);
        fused_layer_kernel<<<grid, 256, 0, stream>>>(x, colOff, ctx16,
                                                     wlp[i], wgp[i], wbp[i],
                                                     pbl, pbg, pbb, out, c);
        colOff += c;
    }
}
